// Block_20693152432548
// MI455X (gfx1250) — matrix operand order repaired and checked by reading, compiled, not run on hardware
//
#include <hip/hip_runtime.h>
#include <hip/hip_bf16.h>
#include <math.h>

#define N_NODES 4096
#define EMBED   512
#define HID     2048
#define NNZ     131072
#define RMS_EPS 1.1920929e-07f

typedef __attribute__((ext_vector_type(16))) __bf16 bf16x16;
typedef __attribute__((ext_vector_type(8)))  __bf16 bf16x8;
typedef __attribute__((ext_vector_type(8)))  float  f32x8;

// ---------------------------------------------------------------------------
// WMMA fragment loaders (16-bit A 16x32 / B 32x16), contiguous k order: k = 16*(lane/16) + s
// Pointers are pre-adjusted for lane (row + hi16); only k0 varies in-loop
// so the unroller can fold it into the vmem instruction-offset field.
// ---------------------------------------------------------------------------
__device__ __forceinline__ bf16x16 load_a(const __bf16* __restrict__ p, int k0) {
  bf16x16 r = *(const bf16x16*)(p + k0);
  return r;
}

__device__ __forceinline__ bf16x16 load_b(const __bf16* __restrict__ p, int k0) {
  return *(const bf16x16*)(p + k0);
}

// ---------------------------------------------------------------------------
// GEMM: C[M,N] = A[M,K](bf16 rowmaj) x B (BT = N x K bf16 rowmaj), f32 accum.
// Epilogue: +bias[n], exact GELU, +res[m,n]; store f32 (Cf) and/or bf16 (Cb).
// Block = 128 thr (4 waves, 2x2); wave tile 32x64 = 2x4 WMMA frags
// -> block tile 64x128, 8 WMMAs per 32-wide k-step per wave.
// Plain loop + unroll 2: compiler pipelines loads across steps with fresh
// registers (no rotation moves, no WMMA->VALU hazard NOPs).
// ---------------------------------------------------------------------------
__global__ __launch_bounds__(128)
void gemm_bf16_wmma(const __bf16* __restrict__ A, const __bf16* __restrict__ BT,
                    float* __restrict__ Cf, __bf16* __restrict__ Cb,
                    const float* __restrict__ bias, const float* __restrict__ res,
                    int M, int N, int K, int doGelu) {
  const int lane = threadIdx.x & 31;
  const int wave = threadIdx.x >> 5;          // 0..3
  const int m0 = blockIdx.y * 64  + (wave >> 1) * 32;
  const int n0 = blockIdx.x * 128 + (wave & 1) * 64;
  const int ml = lane & 15;
  const int hi8  = (lane >> 4) * 8;
  const int hi16 = (lane >> 4) * 16;

  // lane-adjusted base pointers (constant across the k-loop)
  const __bf16* pa0 = A + (size_t)(m0 + ml) * (size_t)K + hi16;
  const __bf16* pa1 = pa0 + (size_t)16 * (size_t)K;
  const __bf16* pb0 = BT + (size_t)(n0 + ml) * (size_t)K + hi16;
  const __bf16* pb1 = pb0 + (size_t)16 * (size_t)K;
  const __bf16* pb2 = pb0 + (size_t)32 * (size_t)K;
  const __bf16* pb3 = pb0 + (size_t)48 * (size_t)K;

  f32x8 acc[2][4] = {};
#pragma unroll 2
  for (int k0 = 0; k0 < K; k0 += 32) {
    bf16x16 aF[2], bF[4];
    aF[0] = load_a(pa0, k0);
    aF[1] = load_a(pa1, k0);
    bF[0] = load_b(pb0, k0);
    bF[1] = load_b(pb1, k0);
    bF[2] = load_b(pb2, k0);
    bF[3] = load_b(pb3, k0);
#pragma unroll
    for (int mi = 0; mi < 2; ++mi)
#pragma unroll
      for (int ni = 0; ni < 4; ++ni)
        acc[mi][ni] = __builtin_amdgcn_wmma_f32_16x16x32_bf16(
            false, aF[mi], false, bF[ni], (short)0, acc[mi][ni], false, false);
  }

  // C/D layout: lane -> N = lane&15; vgpr r -> M = r + 8*(lane>=16)
  const int nl = lane & 15;
  const int mh = (lane >> 4) * 8;
#pragma unroll
  for (int mi = 0; mi < 2; ++mi)
#pragma unroll
    for (int ni = 0; ni < 4; ++ni) {
      const int nCol = n0 + ni * 16 + nl;
#pragma unroll
      for (int r = 0; r < 8; ++r) {
        const int mRow = m0 + mi * 16 + mh + r;
        float v = acc[mi][ni][r];
        if (bias) v += bias[nCol];
        if (doGelu) v = 0.5f * v * (1.0f + erff(v * 0.70710678118654752f));
        const size_t idx = (size_t)mRow * (size_t)N + (size_t)nCol;
        if (res) v += res[idx];
        if (Cf) Cf[idx] = v;
        if (Cb) Cb[idx] = (__bf16)v;
      }
    }
}

// ---------------------------------------------------------------------------
// RMSNorm (row length 512): out = x * rsqrt(mean(x^2)+eps) * g [+ x]
// Writes f32 and/or bf16.
// ---------------------------------------------------------------------------
__global__ __launch_bounds__(256)
void rmsnorm_kernel(const float* __restrict__ x, const float* __restrict__ g,
                    float* __restrict__ outF, __bf16* __restrict__ outB,
                    int addResidual) {
  const int row = blockIdx.x;
  const int t = threadIdx.x;
  const float* xr = x + (size_t)row * EMBED;
  __shared__ float red[8];
  float v0 = xr[t], v1 = xr[t + 256];
  float ss = v0 * v0 + v1 * v1;
#pragma unroll
  for (int off = 16; off; off >>= 1) ss += __shfl_down(ss, off, 32);
  if ((t & 31) == 0) red[t >> 5] = ss;
  __syncthreads();
  if (t == 0) {
    float tot = 0.f;
#pragma unroll
    for (int i = 0; i < 8; ++i) tot += red[i];
    red[0] = tot;
  }
  __syncthreads();
  const float rinv = rsqrtf(red[0] * (1.0f / (float)EMBED) + RMS_EPS);
  float o0 = v0 * rinv * g[t]        + (addResidual ? v0 : 0.f);
  float o1 = v1 * rinv * g[t + 256]  + (addResidual ? v1 : 0.f);
  const size_t base = (size_t)row * EMBED;
  if (outF) { outF[base + t] = o0; outF[base + t + 256] = o1; }
  if (outB) { outB[base + t] = (__bf16)o0; outB[base + t + 256] = (__bf16)o1; }
}

// ---------------------------------------------------------------------------
// Per-node 8-head attention over the head axis (d=64, H=8), acc += attn(q,k,v)
// ---------------------------------------------------------------------------
__global__ __launch_bounds__(512)
void heads_attn_kernel(const float* __restrict__ Q, const float* __restrict__ Km,
                       const float* __restrict__ V, float* __restrict__ acc) {
  const int n = blockIdx.x;
  const int t = threadIdx.x;          // 0..511
  __shared__ float sq[EMBED], sk[EMBED], sv[EMBED], sS[64], sA[64];
  const size_t base = (size_t)n * EMBED;
  sq[t] = Q[base + t]; sk[t] = Km[base + t]; sv[t] = V[base + t];
  __syncthreads();
  if (t < 64) {                        // t = h*8 + g : score(h,g)
    const int h = t >> 3, gg = t & 7;
    float d = 0.f;
#pragma unroll 8
    for (int i = 0; i < 64; ++i) d += sq[h * 64 + i] * sk[gg * 64 + i];
    sS[t] = d * 0.125f;                // 1/sqrt(64)
  }
  __syncthreads();
  if (t < 64) {                        // softmax over g (8 logits per head)
    const int h = t >> 3;
    float mx = -1e30f;
#pragma unroll
    for (int gg = 0; gg < 8; ++gg) mx = fmaxf(mx, sS[h * 8 + gg]);
    float den = 0.f;
#pragma unroll
    for (int gg = 0; gg < 8; ++gg) den += expf(sS[h * 8 + gg] - mx);
    sA[t] = expf(sS[t] - mx) / den;
  }
  __syncthreads();
  const int h = t >> 6, dd = t & 63;   // output element (h, dd)
  float o = 0.f;
#pragma unroll
  for (int gg = 0; gg < 8; ++gg) o += sA[h * 8 + gg] * sv[gg * 64 + dd];
  acc[base + t] += o;
}

// ---------------------------------------------------------------------------
// Sparse adjacency matmul: out[src] += val * pre[dst]   (one block per edge)
// ---------------------------------------------------------------------------
__global__ __launch_bounds__(128)
void segment_sum_kernel(const float* __restrict__ pre, const int* __restrict__ src,
                        const int* __restrict__ dst, const float* __restrict__ val,
                        float* __restrict__ out) {
  const int e = blockIdx.x;
  const int s = src[e], d = dst[e];
  const float v = val[e];
  const float* pr = pre + (size_t)d * EMBED;
  float* po = out + (size_t)s * EMBED;
  for (int c = threadIdx.x; c < EMBED; c += 128)
    atomicAdd(&po[c], v * pr[c]);
}

// ---------------------------------------------------------------------------
// Transposing f32 -> bf16 converter: out[c*R + r] = in[r*C + c]   (R,C %32==0)
// ---------------------------------------------------------------------------
__global__ __launch_bounds__(256)
void transpose_f32_to_bf16(const float* __restrict__ in, __bf16* __restrict__ out,
                           int R, int C) {
  __shared__ float tile[32][33];
  const int c0 = blockIdx.x * 32, r0 = blockIdx.y * 32;
  const int tx = threadIdx.x, ty = threadIdx.y;       // 32 x 8
#pragma unroll
  for (int i = 0; i < 32; i += 8)
    tile[ty + i][tx] = in[(size_t)(r0 + ty + i) * C + (c0 + tx)];
  __syncthreads();
#pragma unroll
  for (int i = 0; i < 32; i += 8)
    out[(size_t)(c0 + ty + i) * R + (r0 + tx)] = (__bf16)tile[tx][ty + i];
}

__global__ __launch_bounds__(256)
void convert_f32_to_bf16_vec4(const float* __restrict__ in, __bf16* __restrict__ out,
                              size_t n4) {
  const size_t i = (size_t)blockIdx.x * blockDim.x + threadIdx.x;
  if (i < n4) {
    float4 v = ((const float4*)in)[i];
    out[4 * i + 0] = (__bf16)v.x; out[4 * i + 1] = (__bf16)v.y;
    out[4 * i + 2] = (__bf16)v.z; out[4 * i + 3] = (__bf16)v.w;
  }
}

__global__ __launch_bounds__(256)
void zero_f32_kernel(float* __restrict__ p, size_t n) {
  const size_t i = (size_t)blockIdx.x * blockDim.x + threadIdx.x;
  if (i < n) p[i] = 0.f;
}

// ---------------------------------------------------------------------------
// Host-side launcher
// ---------------------------------------------------------------------------
static inline void launch_gemm(const __bf16* A, const __bf16* BT, float* Cf, __bf16* Cb,
                               const float* bias, const float* res,
                               int M, int N, int K, int gelu, hipStream_t s) {
  dim3 grid(N / 128, M / 64), block(128);
  gemm_bf16_wmma<<<grid, block, 0, s>>>(A, BT, Cf, Cb, bias, res, M, N, K, gelu);
}

extern "C" void kernel_launch(void* const* d_in, const int* in_sizes, int n_in,
                              void* d_out, int out_size, void* d_ws, size_t ws_size,
                              hipStream_t stream) {
  (void)in_sizes; (void)n_in; (void)out_size; (void)ws_size;
  const float* x_node   = (const float*)d_in[0];
  const float* x_edge   = (const float*)d_in[1];
  const int*   edge_src = (const int*)  d_in[2];
  const int*   edge_dst = (const int*)  d_in[3];
  const float* edge_val = (const float*)d_in[4];
  const float* W[12];
  for (int i = 0; i < 12; ++i) W[i] = (const float*)d_in[5 + i];
  const float* proj_eh = (const float*)d_in[17];
  const float* proj_he = (const float*)d_in[18];
  const float* g1h = (const float*)d_in[19];
  const float* g1e = (const float*)d_in[20];
  const float* g2h = (const float*)d_in[21];
  const float* g2e = (const float*)d_in[22];
  const float* ffh_w1 = (const float*)d_in[23]; const float* ffh_b1 = (const float*)d_in[24];
  const float* ffh_w2 = (const float*)d_in[25]; const float* ffh_b2 = (const float*)d_in[26];
  const float* ffe_w1 = (const float*)d_in[27]; const float* ffe_b1 = (const float*)d_in[28];
  const float* ffe_w2 = (const float*)d_in[29]; const float* ffe_b2 = (const float*)d_in[30];

  const size_t NB = (size_t)N_NODES * EMBED;     // 2,097,152

  // ---- workspace layout ----
  size_t off = 0;
  auto alloc = [&](size_t bytes) -> void* {
    void* p = (char*)d_ws + off;
    off += (bytes + 255) & ~(size_t)255;
    return p;
  };
  float* h       = (float*)alloc(NB * 4);
  float* e       = (float*)alloc(NB * 4);
  float* q_pre   = (float*)alloc(NB * 4);
  float* q_hh    = (float*)alloc(NB * 4);
  float* k_hh    = (float*)alloc(NB * 4);
  float* v_hh    = (float*)alloc(NB * 4);
  float* qe      = (float*)alloc(NB * 4);
  float* k_eh    = (float*)alloc(NB * 4);
  float* v_eh    = (float*)alloc(NB * 4);
  float* q_ee    = (float*)alloc(NB * 4);
  float* k_ee    = (float*)alloc(NB * 4);
  float* v_ee    = (float*)alloc(NB * 4);
  float* qh2     = (float*)alloc(NB * 4);
  float* k_he    = (float*)alloc(NB * 4);
  float* v_he    = (float*)alloc(NB * 4);
  float* qe_proj = (float*)alloc(NB * 4);
  float* qh2_proj= (float*)alloc(NB * 4);
  __bf16* h_bf   = (__bf16*)alloc(NB * 2);
  __bf16* e_bf   = (__bf16*)alloc(NB * 2);
  __bf16* wT[12];
  for (int i = 0; i < 12; ++i) wT[i] = (__bf16*)alloc((size_t)EMBED * EMBED * 2);
  __bf16* w1hT   = (__bf16*)alloc((size_t)HID * EMBED * 2);   // 2048 x 512
  __bf16* w2hT   = (__bf16*)alloc((size_t)EMBED * HID * 2);   // 512 x 2048
  __bf16* w1eT   = (__bf16*)alloc((size_t)HID * EMBED * 2);
  __bf16* w2eT   = (__bf16*)alloc((size_t)EMBED * HID * 2);
  __bf16* projA_eh = (__bf16*)alloc((size_t)N_NODES * N_NODES * 2);
  __bf16* projA_he = (__bf16*)alloc((size_t)N_NODES * N_NODES * 2);
  __bf16* qeT    = (__bf16*)alloc(NB * 2);                    // 512 x 4096
  __bf16* qh2T   = (__bf16*)alloc(NB * 2);
  __bf16* mid_bf = (__bf16*)alloc((size_t)N_NODES * HID * 2); // 4096 x 2048

  float* out_h = (float*)d_out;
  float* out_e = (float*)d_out + NB;

  // ---- 1) pre-norm + residual, bf16 copies ----
  rmsnorm_kernel<<<N_NODES, 256, 0, stream>>>(x_node, g1h, h, h_bf, 1);
  rmsnorm_kernel<<<N_NODES, 256, 0, stream>>>(x_edge, g1e, e, e_bf, 1);

  // ---- 2) weight conversions (transposed bf16: BT = N x K) ----
  {
    dim3 blk(32, 8);
    dim3 gW(EMBED / 32, EMBED / 32);
    for (int i = 0; i < 12; ++i)
      transpose_f32_to_bf16<<<gW, blk, 0, stream>>>(W[i], wT[i], EMBED, EMBED);
    transpose_f32_to_bf16<<<dim3(HID / 32, EMBED / 32), blk, 0, stream>>>(ffh_w1, w1hT, EMBED, HID);
    transpose_f32_to_bf16<<<dim3(EMBED / 32, HID / 32), blk, 0, stream>>>(ffh_w2, w2hT, HID, EMBED);
    transpose_f32_to_bf16<<<dim3(HID / 32, EMBED / 32), blk, 0, stream>>>(ffe_w1, w1eT, EMBED, HID);
    transpose_f32_to_bf16<<<dim3(EMBED / 32, HID / 32), blk, 0, stream>>>(ffe_w2, w2eT, HID, EMBED);
  }
  {
    const size_t n4 = (size_t)N_NODES * N_NODES / 4;
    const int blocks = (int)((n4 + 255) / 256);
    convert_f32_to_bf16_vec4<<<blocks, 256, 0, stream>>>(proj_eh, projA_eh, n4);
    convert_f32_to_bf16_vec4<<<blocks, 256, 0, stream>>>(proj_he, projA_he, n4);
  }

  // ---- 3) twelve 4096x512x512 GEMMs ----
  launch_gemm(h_bf, wT[0], q_pre, nullptr, nullptr, nullptr, N_NODES, EMBED, EMBED, 0, stream);
  launch_gemm(h_bf, wT[1], k_hh,  nullptr, nullptr, nullptr, N_NODES, EMBED, EMBED, 0, stream);
  launch_gemm(h_bf, wT[2], v_hh,  nullptr, nullptr, nullptr, N_NODES, EMBED, EMBED, 0, stream);
  launch_gemm(e_bf, wT[3], qe,    nullptr, nullptr, nullptr, N_NODES, EMBED, EMBED, 0, stream);
  launch_gemm(h_bf, wT[4], k_eh,  nullptr, nullptr, nullptr, N_NODES, EMBED, EMBED, 0, stream);
  launch_gemm(h_bf, wT[5], v_eh,  nullptr, nullptr, nullptr, N_NODES, EMBED, EMBED, 0, stream);
  launch_gemm(e_bf, wT[6], q_ee,  nullptr, nullptr, nullptr, N_NODES, EMBED, EMBED, 0, stream);
  launch_gemm(e_bf, wT[7], k_ee,  nullptr, nullptr, nullptr, N_NODES, EMBED, EMBED, 0, stream);
  launch_gemm(e_bf, wT[8], v_ee,  nullptr, nullptr, nullptr, N_NODES, EMBED, EMBED, 0, stream);
  launch_gemm(h_bf, wT[9], qh2,   nullptr, nullptr, nullptr, N_NODES, EMBED, EMBED, 0, stream);
  launch_gemm(e_bf, wT[10], k_he, nullptr, nullptr, nullptr, N_NODES, EMBED, EMBED, 0, stream);
  launch_gemm(e_bf, wT[11], v_he, nullptr, nullptr, nullptr, N_NODES, EMBED, EMBED, 0, stream);

  // ---- 4) sparse adjacency matmul (segment-sum) ----
  zero_f32_kernel<<<(int)((NB + 255) / 256), 256, 0, stream>>>(q_hh, NB);
  segment_sum_kernel<<<NNZ, 128, 0, stream>>>(q_pre, edge_src, edge_dst, edge_val, q_hh);

  // ---- 5) shape-proj GEMMs (K = 4096) ----
  {
    dim3 blk(32, 8);
    transpose_f32_to_bf16<<<dim3(EMBED / 32, N_NODES / 32), blk, 0, stream>>>(qe,  qeT,  N_NODES, EMBED);
    transpose_f32_to_bf16<<<dim3(EMBED / 32, N_NODES / 32), blk, 0, stream>>>(qh2, qh2T, N_NODES, EMBED);
  }
  launch_gemm(projA_eh, qeT,  qe_proj,  nullptr, nullptr, nullptr, N_NODES, EMBED, N_NODES, 0, stream);
  launch_gemm(projA_he, qh2T, qh2_proj, nullptr, nullptr, nullptr, N_NODES, EMBED, N_NODES, 0, stream);

  // ---- 6) head attentions, accumulated into h / e (residual already inside) ----
  heads_attn_kernel<<<N_NODES, 512, 0, stream>>>(q_hh,     k_hh, v_hh, h);
  heads_attn_kernel<<<N_NODES, 512, 0, stream>>>(qe_proj,  k_eh, v_eh, h);
  heads_attn_kernel<<<N_NODES, 512, 0, stream>>>(q_ee,     k_ee, v_ee, e);
  heads_attn_kernel<<<N_NODES, 512, 0, stream>>>(qh2_proj, k_he, v_he, e);

  // ---- 7) post-norm (bf16 only; residual handled in GEMM2 epilogue) ----
  rmsnorm_kernel<<<N_NODES, 256, 0, stream>>>(h, g2h, nullptr, h_bf, 0);
  rmsnorm_kernel<<<N_NODES, 256, 0, stream>>>(e, g2e, nullptr, e_bf, 0);

  // ---- 8) FFNs: gelu(A W1 + b1) W2 + b2 + residual -> d_out ----
  launch_gemm(h_bf, w1hT, nullptr, mid_bf, ffh_b1, nullptr, N_NODES, HID, EMBED, 1, stream);
  launch_gemm(mid_bf, w2hT, out_h, nullptr, ffh_b2, h, N_NODES, EMBED, HID, 0, stream);
  launch_gemm(e_bf, w1eT, nullptr, mid_bf, ffe_b1, nullptr, N_NODES, HID, EMBED, 1, stream);
  launch_gemm(mid_bf, w2eT, out_e, nullptr, ffe_b2, e, N_NODES, EMBED, HID, 0, stream);
}
